// BigBirdAttention_44856638440215
// MI455X (gfx1250) — compile-verified, matrix operand order checked by reading, not run on hardware
//
#include <hip/hip_runtime.h>
#include <hip/hip_bf16.h>
#include <stdint.h>

// ---------------- problem constants ----------------
#define DM    1024          // d_model
#define NH    16            // heads
#define DKV   64            // head dim (K and V)
#define TBLK  128           // local attention block
#define NBLK  64            // 8192 / 128 blocks
#define NG    16            // global tokens
#define TSEQ  8208          // NG + 8192
#define NBAT  2
#define MTOT  (NBAT * TSEQ) // 16416 rows (multiple of 32)
#define SCL   0.125f        // 1/sqrt(64)

typedef __bf16 bf16_t;
typedef __attribute__((ext_vector_type(16))) bf16_t v16bf;
typedef __attribute__((ext_vector_type(8)))  bf16_t v8bf;
typedef __attribute__((ext_vector_type(8)))  float  v8f;

static __device__ __forceinline__ v8f v8f_zero() {
  v8f z = {0.f, 0.f, 0.f, 0.f, 0.f, 0.f, 0.f, 0.f};
  return z;
}

static __device__ __forceinline__ v8f wmma_bf16(v16bf a, v16bf b, v8f c) {
  // D = A(16x32) * B(32x16) + C, f32 accumulate
  return __builtin_amdgcn_wmma_f32_16x16x32_bf16(false, a, false, b, (short)0, c,
                                                 false, false);
}

// 16x32 bf16 fragment, A-layout (ISA 7.12.2): lane holds row (lane&15);
// elements e<8 -> k = 8*half + e ; e>=8 -> k = 16 + 8*half + (e-8).
// Identical layout serves B-operands sourced from row-major [N,K] matrices
// (B column n == matrix row n). base points at element [row0][k0].
static __device__ __forceinline__ v16bf ld_frag(const bf16_t* base, int stride) {
  const int lane = threadIdx.x & 31;
  const bf16_t* p = base + (lane & 15) * stride + ((lane >> 4) << 3);
  v8bf lo = *(const v8bf*)(p);
  v8bf hi = *(const v8bf*)(p + 16);
  return __builtin_shufflevector(lo, hi, 0, 1, 2, 3, 4, 5, 6, 7, 8, 9, 10, 11,
                                 12, 13, 14, 15);
}

// B-fragment from a row-major [K, N] matrix (per-lane column gather).
static __device__ __forceinline__ v16bf ld_frag_cm(const bf16_t* base, int stride) {
  const int lane = threadIdx.x & 31;
  const int n  = lane & 15;
  const int hh = (lane >> 4) << 3;
  v16bf f;
#pragma unroll
  for (int e = 0; e < 8; ++e) f[e] = base[(hh + e) * stride + n];
#pragma unroll
  for (int e = 0; e < 8; ++e) f[e + 8] = base[(16 + hh + e) * stride + n];
  return f;
}

// Async 16-byte global -> LDS copy (CDNA5 GLOBAL_LOAD_ASYNC_TO_LDS_B128,
// tracked by ASYNCcnt). Per-lane: LDS[lds_off] = MEM[gaddr], 16 bytes.
static __device__ __forceinline__ void async_g2l_b128(uint32_t lds_off,
                                                      uint64_t gaddr) {
  asm volatile("global_load_async_to_lds_b128 %0, %1, off"
               :: "v"(lds_off), "v"(gaddr)
               : "memory");
}
static __device__ __forceinline__ void wait_async0() {
  asm volatile("s_wait_asynccnt 0x0" ::: "memory");
}

// ---------------- kernel 1: f32 -> bf16 convert (8 elems/thread) -------------
__global__ __launch_bounds__(256) void k_cvt(const float* __restrict__ s,
                                             bf16_t* __restrict__ d, int nel) {
  int i = (blockIdx.x * 256 + threadIdx.x) * 8;
  if (i + 8 <= nel) {
    v8f x = *(const v8f*)(s + i);
    v8bf y;
#pragma unroll
    for (int e = 0; e < 8; ++e) y[e] = (bf16_t)x[e];
    *(v8bf*)(d + i) = y;
  } else {
    for (; i < nel; ++i) d[i] = (bf16_t)s[i];
  }
}

// ---------------- kernel 2: QKV projection GEMM ----------------
// Y[m,n] = sum_k x[m,k] * W[n,k]; one wave computes a 32x64 tile.
// K loop manually unrolled 2x into a ping-pong so the two fragment buffers
// get disjoint registers (no rotation copies, fewer WMMA->VALU hazard nops).
__global__ __launch_bounds__(32) void k_proj(
    const bf16_t* __restrict__ xb, const bf16_t* __restrict__ wq,
    const bf16_t* __restrict__ wk, const bf16_t* __restrict__ wv,
    bf16_t* __restrict__ Qb, bf16_t* __restrict__ Kb, bf16_t* __restrict__ Vb) {
  const bf16_t* W = blockIdx.z == 0 ? wq : (blockIdx.z == 1 ? wk : wv);
  bf16_t* dst = blockIdx.z == 0 ? Qb : (blockIdx.z == 1 ? Kb : Vb);
  const int mbase = blockIdx.x * 32;
  const int nbase = blockIdx.y * 64;

  v8f acc[2][4];
#pragma unroll
  for (int i = 0; i < 2; ++i)
#pragma unroll
    for (int c = 0; c < 4; ++c) acc[i][c] = v8f_zero();

  const bf16_t* arow0 = xb + (size_t)mbase * DM;

  auto ldset = [&](int kt, v16bf& A0, v16bf& A1, v16bf* B) {
    A0 = ld_frag(arow0 + kt, DM);
    A1 = ld_frag(arow0 + (size_t)16 * DM + kt, DM);
#pragma unroll
    for (int c = 0; c < 4; ++c)
      B[c] = ld_frag(W + (size_t)(nbase + c * 16) * DM + kt, DM);
  };
  auto mul = [&](const v16bf& A0, const v16bf& A1, const v16bf* B) {
#pragma unroll
    for (int c = 0; c < 4; ++c) {
      acc[0][c] = wmma_bf16(A0, B[c], acc[0][c]);
      acc[1][c] = wmma_bf16(A1, B[c], acc[1][c]);
    }
  };

  v16bf A0a, A1a, Ba[4], A0b, A1b, Bb[4];
  ldset(0, A0a, A1a, Ba);
  for (int kt = 0; kt < DM; kt += 64) {
    ldset(kt + 32, A0b, A1b, Bb);
    mul(A0a, A1a, Ba);
    if (kt + 64 < DM) ldset(kt + 64, A0a, A1a, Ba);
    mul(A0b, A1b, Bb);
  }

  const int lane = threadIdx.x & 31, n = lane & 15, half = lane >> 4;
#pragma unroll
  for (int mt = 0; mt < 2; ++mt)
#pragma unroll
    for (int c = 0; c < 4; ++c)
#pragma unroll
      for (int r = 0; r < 8; ++r) {
        int row = mbase + mt * 16 + r + 8 * half;
        int col = nbase + c * 16 + n;
        int bb = row >= TSEQ ? 1 : 0;
        int t = row - bb * TSEQ;
        dst[(((size_t)bb * NH + (col >> 6)) * TSEQ + t) * DKV + (col & 63)] =
            (bf16_t)acc[mt][c][r];
      }
}

// ---------------- kernel 3a: local block attention ----------------
#define SVT 392  // Vt LDS stride (elems): 784B rows, 16B aligned, conflict-free
#define SKL 72   // Klds stride (elems): 144B rows, 16B aligned, conflict-free
__global__ __launch_bounds__(256) void k_attn_local(
    const bf16_t* __restrict__ Qb, const bf16_t* __restrict__ Kb,
    const bf16_t* __restrict__ Vb, bf16_t* __restrict__ attnb) {
  __shared__ __align__(16) bf16_t Klds[384 * SKL];    // 55296 B rolled K window
  __shared__ __align__(16) bf16_t Vt[64 * SVT];       // 50176 B V window transposed
  __shared__ __align__(16) bf16_t Vgt[64 * 32];       // 4096 B padded global V^T
  __shared__ __align__(16) bf16_t Ppad[8 * 16 * 40];  // 10240 B per-wave P staging

  const int nb = blockIdx.x, h = blockIdx.y, b = blockIdx.z;
  const int tid = threadIdx.x;
  const size_t bh = (size_t)(b * NH + h) * TSEQ;

  // ---- stage rolled K window via async global->LDS B128 copies ----
  // 384 rows x 64 bf16 = 3072 16B chunks, 12 per thread.
  for (int idx = tid; idx < 384 * 8; idx += 256) {
    int row = idx >> 3, ch = idx & 7;
    int blk = nb + (row >> 7) - 1;
    blk = (blk + NBLK) % NBLK;
    const bf16_t* gp =
        Kb + (bh + NG + blk * TBLK + (row & 127)) * DKV + ch * 8;
    async_g2l_b128((uint32_t)(uintptr_t)&Klds[row * SKL + ch * 8],
                   (uint64_t)(uintptr_t)gp);
  }

  // stage V window (transposed, with circular roll) and padded global V
  for (int w2 = tid; w2 < 3 * TBLK; w2 += 256) {
    int blk = nb + (w2 >> 7) - 1;
    blk = (blk + NBLK) % NBLK;
    const bf16_t* src = Vb + (bh + NG + blk * TBLK + (w2 & 127)) * DKV;
#pragma unroll
    for (int d = 0; d < DKV; ++d) Vt[d * SVT + w2] = src[d];
  }
  if (tid < 32) {
    if (tid < NG) {
      const bf16_t* src = Vb + (bh + tid) * DKV;
#pragma unroll
      for (int d = 0; d < DKV; ++d) Vgt[d * 32 + tid] = src[d];
    } else {
#pragma unroll
      for (int d = 0; d < DKV; ++d) Vgt[d * 32 + tid] = (bf16_t)0.f;
    }
  }
  wait_async0();
  __syncthreads();

  const int w = tid >> 5, lane = tid & 31;
  const int n = lane & 15, half = lane >> 4;
  bf16_t* pp = &Ppad[w * 640];

  const int qrow = NG + nb * TBLK + w * 16;
  const bf16_t* qbase = Qb + (bh + qrow) * DKV;
  v16bf a0 = ld_frag(qbase, DKV);
  v16bf a1 = ld_frag(qbase + 32, DKV);

  // ---- global-key scores (separate softmax over 16 keys) ----
  v8f sg;
  {
    const bf16_t* kgb = Kb + bh * DKV;  // K rows t=0..15
    v16bf b0 = ld_frag(kgb, DKV);
    v16bf b1 = ld_frag(kgb + 32, DKV);
    sg = wmma_bf16(a0, b0, v8f_zero());
    sg = wmma_bf16(a1, b1, sg);
  }
  float mg[8], lg[8];
#pragma unroll
  for (int r = 0; r < 8; ++r) { mg[r] = sg[r] * SCL; lg[r] = 1.f; }
#pragma unroll
  for (int off = 1; off < 16; off <<= 1)
#pragma unroll
    for (int r = 0; r < 8; ++r) {
      float om = __shfl_xor(mg[r], off, 16);
      float ol = __shfl_xor(lg[r], off, 16);
      float nm = fmaxf(mg[r], om);
      lg[r] = lg[r] * __expf(mg[r] - nm) + ol * __expf(om - nm);
      mg[r] = nm;
    }

  // ---- local window pass 1: streaming row max / sum (K from LDS) ----
  float mx[8], lx[8];
#pragma unroll
  for (int r = 0; r < 8; ++r) { mx[r] = -1e30f; lx[r] = 0.f; }
  for (int j = 0; j < 24; ++j) {
    const bf16_t* kb2 = &Klds[(j * 16) * SKL];
    v16bf b0 = ld_frag(kb2, SKL);
    v16bf b1 = ld_frag(kb2 + 32, SKL);
    v8f s = wmma_bf16(a0, b0, v8f_zero());
    s = wmma_bf16(a1, b1, s);
#pragma unroll
    for (int r = 0; r < 8; ++r) {
      float v = s[r] * SCL;
      float nm = fmaxf(mx[r], v);
      lx[r] = lx[r] * __expf(mx[r] - nm) + __expf(v - nm);
      mx[r] = nm;
    }
  }
#pragma unroll
  for (int off = 1; off < 16; off <<= 1)
#pragma unroll
    for (int r = 0; r < 8; ++r) {
      float om = __shfl_xor(mx[r], off, 16);
      float ol = __shfl_xor(lx[r], off, 16);
      float nm = fmaxf(mx[r], om);
      lx[r] = lx[r] * __expf(mx[r] - nm) + ol * __expf(om - nm);
      mx[r] = nm;
    }

  // ---- output accumulation ----
  v8f O[4], Og[4];
#pragma unroll
  for (int c = 0; c < 4; ++c) { O[c] = v8f_zero(); Og[c] = v8f_zero(); }

  // global PV: P (16x16) padded to 16x32 with zeros, Vgt rows 16..31 are zero
#pragma unroll
  for (int r = 0; r < 8; ++r) {
    pp[(r + 8 * half) * 40 + n] = (bf16_t)__expf(sg[r] * SCL - mg[r]);
    pp[(r + 8 * half) * 40 + 16 + n] = (bf16_t)0.f;
  }
  {
    v16bf ap = ld_frag(pp, 40);
#pragma unroll
    for (int c = 0; c < 4; ++c) {
      v16bf bv = ld_frag(&Vgt[(c * 16) * 32], 32);
      Og[c] = wmma_bf16(ap, bv, Og[c]);
    }
  }

  // local pass 2: recompute scores, exp, relayout via LDS, PV
  for (int jp = 0; jp < 12; ++jp) {
#pragma unroll
    for (int jj = 0; jj < 2; ++jj) {
      int j = jp * 2 + jj;
      const bf16_t* kb2 = &Klds[(j * 16) * SKL];
      v16bf b0 = ld_frag(kb2, SKL);
      v16bf b1 = ld_frag(kb2 + 32, SKL);
      v8f s = wmma_bf16(a0, b0, v8f_zero());
      s = wmma_bf16(a1, b1, s);
#pragma unroll
      for (int r = 0; r < 8; ++r)
        pp[(r + 8 * half) * 40 + jj * 16 + n] =
            (bf16_t)__expf(s[r] * SCL - mx[r]);
    }
    v16bf ap = ld_frag(pp, 40);
#pragma unroll
    for (int c = 0; c < 4; ++c) {
      v16bf bv = ld_frag(&Vt[(c * 16) * SVT + jp * 32], SVT);
      O[c] = wmma_bf16(ap, bv, O[c]);
    }
  }

  // ---- store (normalize each softmax separately, then add) ----
  const size_t orow0 = (size_t)b * TSEQ + qrow;
#pragma unroll
  for (int c = 0; c < 4; ++c)
#pragma unroll
    for (int r = 0; r < 8; ++r) {
      size_t row = orow0 + r + 8 * half;
      attnb[row * DM + h * DKV + c * 16 + n] =
          (bf16_t)(O[c][r] / lx[r] + Og[c][r] / lg[r]);
    }
}

// ---------------- kernel 3b: global queries attend all 8192 local keys -------
__global__ __launch_bounds__(256) void k_attn_global(
    const bf16_t* __restrict__ Qb, const bf16_t* __restrict__ Kb,
    const bf16_t* __restrict__ Vb, bf16_t* __restrict__ attnb) {
  __shared__ __align__(16) bf16_t Ppad[8 * 16 * 40];
  __shared__ float smM[8 * 16];
  __shared__ float smL[8 * 16];
  __shared__ float smO[8 * 16 * 64];

  const int h = blockIdx.x, b = blockIdx.y;
  const size_t bh = (size_t)(b * NH + h) * TSEQ;
  const int tid = threadIdx.x, w = tid >> 5, lane = tid & 31;
  const int n = lane & 15, half = lane >> 4;
  bf16_t* pp = &Ppad[w * 640];

  const bf16_t* qbase = Qb + bh * DKV;  // global queries t=0..15
  v16bf a0 = ld_frag(qbase, DKV);
  v16bf a1 = ld_frag(qbase + 32, DKV);

  // phase A: this wave covers keys [w*1024, w*1024+1024)
  float mx[8], lx[8];
#pragma unroll
  for (int r = 0; r < 8; ++r) { mx[r] = -1e30f; lx[r] = 0.f; }
  for (int i = 0; i < 64; ++i) {
    const bf16_t* kb2 = Kb + (bh + NG + (w * 64 + i) * 16) * DKV;
    v16bf b0 = ld_frag(kb2, DKV);
    v16bf b1 = ld_frag(kb2 + 32, DKV);
    v8f s = wmma_bf16(a0, b0, v8f_zero());
    s = wmma_bf16(a1, b1, s);
#pragma unroll
    for (int r = 0; r < 8; ++r) {
      float v = s[r] * SCL;
      float nm = fmaxf(mx[r], v);
      lx[r] = lx[r] * __expf(mx[r] - nm) + __expf(v - nm);
      mx[r] = nm;
    }
  }
#pragma unroll
  for (int off = 1; off < 16; off <<= 1)
#pragma unroll
    for (int r = 0; r < 8; ++r) {
      float om = __shfl_xor(mx[r], off, 16);
      float ol = __shfl_xor(lx[r], off, 16);
      float nm = fmaxf(mx[r], om);
      lx[r] = lx[r] * __expf(mx[r] - nm) + ol * __expf(om - nm);
      mx[r] = nm;
    }
  if (n == 0) {
#pragma unroll
    for (int r = 0; r < 8; ++r) {
      smM[w * 16 + r + 8 * half] = mx[r];
      smL[w * 16 + r + 8 * half] = lx[r];
    }
  }
  __syncthreads();

  // combined row max across waves
  float Mc[8];
#pragma unroll
  for (int r = 0; r < 8; ++r) {
    int m = r + 8 * half;
    float M = -1e30f;
#pragma unroll
    for (int w2 = 0; w2 < 8; ++w2) M = fmaxf(M, smM[w2 * 16 + m]);
    Mc[r] = M;
  }

  // phase B: recompute, exp with combined max, PV partials
  v8f O[4];
#pragma unroll
  for (int c = 0; c < 4; ++c) O[c] = v8f_zero();
  for (int jp = 0; jp < 32; ++jp) {
    int kb0 = w * 1024 + jp * 32;  // key offset in [0,8192)
#pragma unroll
    for (int jj = 0; jj < 2; ++jj) {
      const bf16_t* kb2 = Kb + (bh + NG + kb0 + jj * 16) * DKV;
      v16bf b0 = ld_frag(kb2, DKV);
      v16bf b1 = ld_frag(kb2 + 32, DKV);
      v8f s = wmma_bf16(a0, b0, v8f_zero());
      s = wmma_bf16(a1, b1, s);
#pragma unroll
      for (int r = 0; r < 8; ++r)
        pp[(r + 8 * half) * 40 + jj * 16 + n] =
            (bf16_t)__expf(s[r] * SCL - Mc[r]);
    }
    v16bf ap = ld_frag(pp, 40);
#pragma unroll
    for (int c = 0; c < 4; ++c) {
      v16bf bv = ld_frag_cm(Vb + (bh + NG + kb0) * DKV + c * 16, DKV);
      O[c] = wmma_bf16(ap, bv, O[c]);
    }
  }
#pragma unroll
  for (int c = 0; c < 4; ++c)
#pragma unroll
    for (int r = 0; r < 8; ++r)
      smO[w * 1024 + (r + 8 * half) * 64 + c * 16 + n] = O[c][r];
  __syncthreads();

  // final cross-wave reduction + normalization; 4 outputs per thread
#pragma unroll
  for (int q = 0; q < 4; ++q) {
    int idx = tid * 4 + q;
    int m = idx >> 6, d = idx & 63;
    float M = -1e30f;
#pragma unroll
    for (int w2 = 0; w2 < 8; ++w2) M = fmaxf(M, smM[w2 * 16 + m]);
    float L = 0.f;
#pragma unroll
    for (int w2 = 0; w2 < 8; ++w2)
      L += smL[w2 * 16 + m] * __expf(smM[w2 * 16 + m] - M);
    float acc = 0.f;
#pragma unroll
    for (int w2 = 0; w2 < 8; ++w2) acc += smO[w2 * 1024 + idx];
    attnb[((size_t)b * TSEQ + m) * DM + h * DKV + d] = (bf16_t)(acc / L);
  }
}

// ---------------- kernel 4: output projection + bias (f32 out) ----------------
__global__ __launch_bounds__(32) void k_oproj(const bf16_t* __restrict__ ab,
                                              const bf16_t* __restrict__ wo,
                                              const float* __restrict__ bo,
                                              float* __restrict__ out) {
  const int mbase = blockIdx.x * 32;
  const int nbase = blockIdx.y * 64;
  v8f acc[2][4];
#pragma unroll
  for (int i = 0; i < 2; ++i)
#pragma unroll
    for (int c = 0; c < 4; ++c) acc[i][c] = v8f_zero();

  const bf16_t* arow0 = ab + (size_t)mbase * DM;

  auto ldset = [&](int kt, v16bf& A0, v16bf& A1, v16bf* B) {
    A0 = ld_frag(arow0 + kt, DM);
    A1 = ld_frag(arow0 + (size_t)16 * DM + kt, DM);
#pragma unroll
    for (int c = 0; c < 4; ++c)
      B[c] = ld_frag(wo + (size_t)(nbase + c * 16) * DM + kt, DM);
  };
  auto mul = [&](const v16bf& A0, const v16bf& A1, const v16bf* B) {
#pragma unroll
    for (int c = 0; c < 4; ++c) {
      acc[0][c] = wmma_bf16(A0, B[c], acc[0][c]);
      acc[1][c] = wmma_bf16(A1, B[c], acc[1][c]);
    }
  };

  v16bf A0a, A1a, Ba[4], A0b, A1b, Bb[4];
  ldset(0, A0a, A1a, Ba);
  for (int kt = 0; kt < DM; kt += 64) {
    ldset(kt + 32, A0b, A1b, Bb);
    mul(A0a, A1a, Ba);
    if (kt + 64 < DM) ldset(kt + 64, A0a, A1a, Ba);
    mul(A0b, A1b, Bb);
  }

  const int lane = threadIdx.x & 31, n = lane & 15, half = lane >> 4;
#pragma unroll
  for (int mt = 0; mt < 2; ++mt)
#pragma unroll
    for (int c = 0; c < 4; ++c)
#pragma unroll
      for (int r = 0; r < 8; ++r) {
        int row = mbase + mt * 16 + r + 8 * half;
        int col = nbase + c * 16 + n;
        out[(size_t)row * DM + col] = acc[mt][c][r] + bo[col];
      }
}

// ---------------- host launcher ----------------
extern "C" void kernel_launch(void* const* d_in, const int* in_sizes, int n_in,
                              void* d_out, int out_size, void* d_ws,
                              size_t ws_size, hipStream_t stream) {
  const float* x  = (const float*)d_in[0];
  const float* Wq = (const float*)d_in[1];
  const float* Wk = (const float*)d_in[2];
  const float* Wv = (const float*)d_in[3];
  const float* Wo = (const float*)d_in[4];
  const float* bo = (const float*)d_in[5];

  char* ws = (char*)d_ws;
  size_t off = 0;
  auto take = [&](size_t bytes) -> char* {
    char* p = ws + off;
    off += (bytes + 255) & ~(size_t)255;
    return p;
  };

  bf16_t* xb  = (bf16_t*)take((size_t)MTOT * DM * 2);
  bf16_t* wqb = (bf16_t*)take((size_t)DM * DM * 2);
  bf16_t* wkb = (bf16_t*)take((size_t)DM * DM * 2);
  bf16_t* wvb = (bf16_t*)take((size_t)DM * DM * 2);
  bf16_t* wob = (bf16_t*)take((size_t)DM * DM * 2);
  bf16_t* Qb  = (bf16_t*)take((size_t)NBAT * NH * TSEQ * DKV * 2);
  bf16_t* Kb  = (bf16_t*)take((size_t)NBAT * NH * TSEQ * DKV * 2);
  bf16_t* Vb  = (bf16_t*)take((size_t)NBAT * NH * TSEQ * DKV * 2);
  bf16_t* ab  = (bf16_t*)take((size_t)MTOT * DM * 2);

  const int nx = MTOT * DM;
  const int nw = DM * DM;
  k_cvt<<<(nx / 8 + 255) / 256, 256, 0, stream>>>(x, xb, nx);
  k_cvt<<<(nw / 8 + 255) / 256, 256, 0, stream>>>(Wq, wqb, nw);
  k_cvt<<<(nw / 8 + 255) / 256, 256, 0, stream>>>(Wk, wkb, nw);
  k_cvt<<<(nw / 8 + 255) / 256, 256, 0, stream>>>(Wv, wvb, nw);
  k_cvt<<<(nw / 8 + 255) / 256, 256, 0, stream>>>(Wo, wob, nw);

  k_proj<<<dim3(MTOT / 32, DM / 64, 3), 32, 0, stream>>>(xb, wqb, wkb, wvb, Qb,
                                                         Kb, Vb);
  k_attn_local<<<dim3(NBLK, NH, NBAT), 256, 0, stream>>>(Qb, Kb, Vb, ab);
  k_attn_global<<<dim3(NH, NBAT), 256, 0, stream>>>(Qb, Kb, Vb, ab);
  k_oproj<<<dim3(MTOT / 32, DM / 64), 32, 0, stream>>>(ab, wob, bo,
                                                       (float*)d_out);
}
